// MotionEncoder_55173149885015
// MI455X (gfx1250) — matrix operand order repaired and checked by reading, not compiled, not run on hardware
//
#include <hip/hip_runtime.h>

#define B_   256
#define F_   256
#define NJ3_ 51
#define D_   512
#define TC_  32   // time chunk for XP precompute (64MB f32, fits L2)

typedef __attribute__((ext_vector_type(16))) __bf16 v16bf;
typedef __attribute__((ext_vector_type(8)))  float  v8f;

union Frag { uint4 u[2]; v16bf v; };

__device__ __forceinline__ unsigned short f2bf(float f) {
  unsigned int u = __float_as_uint(f);
  u += 0x7FFFu + ((u >> 16) & 1u);           // round-to-nearest-even
  return (unsigned short)(u >> 16);
}
__device__ __forceinline__ float sigm(float x) { return 1.0f / (1.0f + __expf(-x)); }

// ---------------------------------------------------------------------------
// WMMA GEMM: out[M,N] = A[M,K](bf16,row-major) @ B[N,K](bf16,row-major).T
//            (+ Cin[M,N])  (+ bias[N])  (relu)  -> f32 and/or bf16 outputs
// One wave computes a (MT*16)x64 tile with MT*4 v_wmma_f32_16x16x32_bf16 per
// K-step (A fragments reused across the 4 N-subtiles, B fragments across MT).
// K % 32 == 0, M % (16*MT) == 0, N % 64 == 0. 4 waves / block.
// ---------------------------------------------------------------------------
template<int MT, bool HAS_CIN, bool HAS_BIAS, bool RELU, bool OUT_F32, bool OUT_BF16>
__global__ __launch_bounds__(128) void k_gemm(
    const unsigned short* __restrict__ A, const unsigned short* __restrict__ Bm,
    const float* __restrict__ Cin, const float* __restrict__ bias,
    float* __restrict__ outF, unsigned short* __restrict__ outB,
    int M, int N, int K)
{
  const int wave = threadIdx.x >> 5;
  const int lane = threadIdx.x & 31;
  const int tilesN = N >> 6;
  const int tile = blockIdx.x * 4 + wave;
  const int tM = tile / tilesN;
  const int tN = tile - tM * tilesN;
  if (tM >= M / (16 * MT)) return;            // wave-uniform: EXEC stays all-1s

  const int r    = lane & 15;
  const int hsel = lane >> 4;                 // which half of the wave
  const int half = hsel * 8;
  const size_t row0 = (size_t)tM * 16 * MT;
  const int    col0 = tN * 64;

  v8f acc[MT][4];
  if (HAS_CIN) {
    #pragma unroll
    for (int i = 0; i < MT; ++i)
      #pragma unroll
      for (int j = 0; j < 4; ++j) {
        const float* cp = Cin + (row0 + i * 16 + half) * N + col0 + j * 16 + r;
        #pragma unroll
        for (int e = 0; e < 8; ++e) acc[i][j][e] = cp[(size_t)e * N];
      }
  } else {
    #pragma unroll
    for (int i = 0; i < MT; ++i)
      #pragma unroll
      for (int j = 0; j < 4; ++j)
        #pragma unroll
        for (int e = 0; e < 8; ++e) acc[i][j][e] = 0.0f;
  }

  // A fragment: lanes 0-15 K={k0..k0+7, k0+16..k0+23}; lanes 16-31 shifted by 8
  const unsigned short* ap[MT];
  #pragma unroll
  for (int i = 0; i < MT; ++i) ap[i] = A + (row0 + i * 16 + r) * (size_t)K + half;

  for (int k0 = 0; k0 < K; k0 += 32) {
    Frag a[MT];
    #pragma unroll
    for (int i = 0; i < MT; ++i) {
      a[i].u[0] = *(const uint4*)(ap[i] + k0);
      a[i].u[1] = *(const uint4*)(ap[i] + k0 + 16);
    }
    // speculative prefetch of the A stream a few K-steps ahead (OOB is dropped)
    __builtin_prefetch(ap[0] + k0 + 128, 0, 1);
    #pragma unroll
    for (int j = 0; j < 4; ++j) {
      // B fragment: lanes 0-15 K={k0..k0+7, k0+16..k0+23}; lanes 16-31 shifted by 8
      const unsigned short* bp = Bm + (size_t)(col0 + j * 16 + r) * K + k0 + hsel * 8;
      Frag b;
      b.u[0] = *(const uint4*)bp;
      b.u[1] = *(const uint4*)(bp + 16);
      #pragma unroll
      for (int i = 0; i < MT; ++i)
        acc[i][j] = __builtin_amdgcn_wmma_f32_16x16x32_bf16(
            false, a[i].v, false, b.v, (short)0, acc[i][j], false, false);
    }
  }

  #pragma unroll
  for (int i = 0; i < MT; ++i)
    #pragma unroll
    for (int j = 0; j < 4; ++j) {
      const int col = col0 + j * 16 + r;
      const float bv = HAS_BIAS ? bias[col] : 0.0f;
      #pragma unroll
      for (int e = 0; e < 8; ++e) {
        float v = acc[i][j][e] + bv;
        if (RELU) v = v > 0.0f ? v : 0.0f;
        const size_t row = row0 + i * 16 + e + half;
        if (OUT_F32)  outF[row * N + col] = v;
        if (OUT_BF16) outB[row * N + col] = f2bf(v);
      }
    }
}

// ------------------------- weight / input prep -----------------------------

// dst[n][k] (bf16, [dstN x dstK]) = src[k][n] (f32, [srcK x srcN]), zero-padded
__global__ void k_convT(const float* __restrict__ src, unsigned short* __restrict__ dst,
                        int srcK, int srcN, int dstK, int dstN) {
  int id = blockIdx.x * 256 + threadIdx.x;
  if (id >= dstK * dstN) return;
  int n = id / dstK, k = id - n * dstK;
  float v = (n < srcN && k < srcK) ? src[(size_t)k * srcN + n] : 0.0f;
  dst[id] = f2bf(v);
}

__global__ void k_w0a(const float* __restrict__ Wih0, unsigned short* __restrict__ dst) {
  int id = blockIdx.x * 256 + threadIdx.x;           // 2048*512
  if (id >= 2048 * 512) return;
  int n = id >> 9, c = id & 511;
  dst[id] = f2bf(Wih0[(size_t)n * 563 + c]);
}

__global__ void k_w0cat(const float* __restrict__ Wih0, const float* __restrict__ Whh0,
                        unsigned short* __restrict__ dst) {
  int id = blockIdx.x * 256 + threadIdx.x;           // 2048*576
  if (id >= 2048 * 576) return;
  int n = id / 576, c = id - n * 576;
  float v = (c < 64) ? ((c < NJ3_) ? Wih0[(size_t)n * 563 + 512 + c] : 0.0f)
                     : Whh0[(size_t)n * 512 + (c - 64)];
  dst[id] = f2bf(v);
}

__global__ void k_w1cat(const float* __restrict__ Wih1, const float* __restrict__ Whh1,
                        unsigned short* __restrict__ dst) {
  int id = blockIdx.x * 256 + threadIdx.x;           // 2048*1024
  if (id >= 2048 * 1024) return;
  int n = id >> 10, c = id & 1023;
  float v = (c < 512) ? Wih1[(size_t)n * 512 + c] : Whh1[(size_t)n * 512 + (c - 512)];
  dst[id] = f2bf(v);
}

__global__ void k_bias(const float* bih0, const float* bhh0,
                       const float* bih1, const float* bhh1,
                       const float* dec_b,
                       float* bsum0, float* bsum1, float* decb64) {
  int id = blockIdx.x * 256 + threadIdx.x;
  if (id < 2048) { bsum0[id] = bih0[id] + bhh0[id]; bsum1[id] = bih1[id] + bhh1[id]; }
  if (id < 64)   decb64[id] = (id < NJ3_) ? dec_b[id] : 0.0f;
}

// x (B,F,34) f32  ->  Xbf time-major [f*B+b][64] bf16, K padded 34->64
__global__ void k_xbf(const float* __restrict__ x, unsigned short* __restrict__ dst) {
  int id = blockIdx.x * 256 + threadIdx.x;           // 65536*64
  if (id >= 65536 * 64) return;
  int rm = id >> 6, c = id & 63;
  int f = rm >> 8, b = rm & 255;
  float v = (c < 34) ? x[((size_t)b * F_ + f) * 34 + c] : 0.0f;
  dst[id] = f2bf(v);
}

// init (B,1,85) -> bf16 padded [256 x 96]
__global__ void k_initbf(const float* __restrict__ init, unsigned short* __restrict__ dst) {
  int id = blockIdx.x * 256 + threadIdx.x;           // 256*96
  if (id >= 256 * 96) return;
  int b = id / 96, c = id - b * 96;
  dst[id] = f2bf((c < 85) ? init[(size_t)b * 85 + c] : 0.0f);
}

// pred0 = init[:,0,:51] into inp0 cols 0..63 (pad zero, pad stays zero all scan)
__global__ void k_inp0init(const float* __restrict__ init, unsigned short* __restrict__ inp0) {
  int id = blockIdx.x * 256 + threadIdx.x;           // 256*64
  if (id >= 256 * 64) return;
  int b = id >> 6, c = id & 63;
  inp0[(size_t)b * 576 + c] = f2bf((c < NJ3_) ? init[(size_t)b * 85 + c] : 0.0f);
}

// z3 (256 x 2048) -> h0 -> inp0[:,64:576], h1 -> inp1[:,512:1024], c0, c1
__global__ void k_scatter_init(const float* __restrict__ z3,
                               unsigned short* __restrict__ inp0,
                               unsigned short* __restrict__ inp1,
                               float* __restrict__ c0, float* __restrict__ c1) {
  int id = blockIdx.x * 256 + threadIdx.x;           // 256*512
  if (id >= 256 * 512) return;
  int b = id >> 9, d = id & 511;
  const float* z = z3 + (size_t)b * 2048;
  inp0[(size_t)b * 576 + 64 + d]   = f2bf(z[d]);          // h layer0
  inp1[(size_t)b * 1024 + 512 + d] = f2bf(z[512 + d]);    // h layer1
  c0[id] = z[1024 + d];
  c1[id] = z[1536 + d];
}

// LSTM cell: gates(256x2048 f32) + c -> new c (in place), h as bf16 to 2 dests,
// optionally h as f32 into motion_context
__global__ void k_cell(const float* __restrict__ gates, float* __restrict__ cbuf,
                       unsigned short* __restrict__ o1, int ld1,
                       unsigned short* __restrict__ o2, int ld2,
                       float* __restrict__ ctx, int t) {
  int id = blockIdx.x * 256 + threadIdx.x;           // 256*512
  if (id >= 256 * 512) return;
  int b = id >> 9, d = id & 511;
  const float* g = gates + (size_t)b * 2048;
  float ig = sigm(g[d]);
  float fg = sigm(g[512 + d]);
  float gg = tanhf(g[1024 + d]);
  float og = sigm(g[1536 + d]);
  float cn = fg * cbuf[id] + ig * gg;
  float hn = og * tanhf(cn);
  cbuf[id] = cn;
  unsigned short hb = f2bf(hn);
  o1[(size_t)b * ld1 + d] = hb;
  o2[(size_t)b * ld2 + d] = hb;
  if (ctx) ctx[((size_t)b * F_ + t) * 563 + d] = hn;
}

// pred (256x64 f32) -> pred_kp3d, motion_context[:,512:563], inp0[:,0:51] bf16
__global__ void k_spred(const float* __restrict__ pred,
                        float* __restrict__ outkp, float* __restrict__ outctx,
                        unsigned short* __restrict__ inp0, int t) {
  int id = blockIdx.x * 256 + threadIdx.x;           // 256*51
  if (id >= 256 * NJ3_) return;
  int b = id / NJ3_, c = id - b * NJ3_;
  float v = pred[(size_t)b * 64 + c];
  outkp[((size_t)b * F_ + t) * NJ3_ + c] = v;
  outctx[((size_t)b * F_ + t) * 563 + 512 + c] = v;
  inp0[(size_t)b * 576 + c] = f2bf(v);
}

// ---------------------------------------------------------------------------

#define EW(kern, n, ...) kern<<<((n) + 255) / 256, 256, 0, stream>>>(__VA_ARGS__)
#define GEMM_GRID(M, N, MT) ((((M) / (16 * (MT))) * ((N) >> 6) + 3) / 4)

extern "C" void kernel_launch(void* const* d_in, const int* in_sizes, int n_in,
                              void* d_out, int out_size, void* d_ws, size_t ws_size,
                              hipStream_t stream) {
  const float* x       = (const float*)d_in[0];
  const float* init_   = (const float*)d_in[1];
  const float* embed_W = (const float*)d_in[2];
  const float* embed_b = (const float*)d_in[3];
  const float* ni_W1   = (const float*)d_in[4];
  const float* ni_b1   = (const float*)d_in[5];
  const float* ni_W2   = (const float*)d_in[6];
  const float* ni_b2   = (const float*)d_in[7];
  const float* ni_W3   = (const float*)d_in[8];
  const float* ni_b3   = (const float*)d_in[9];
  const float* Wih0    = (const float*)d_in[10];
  const float* Whh0    = (const float*)d_in[11];
  const float* bih0    = (const float*)d_in[12];
  const float* bhh0    = (const float*)d_in[13];
  const float* Wih1    = (const float*)d_in[14];
  const float* Whh1    = (const float*)d_in[15];
  const float* bih1    = (const float*)d_in[16];
  const float* bhh1    = (const float*)d_in[17];
  const float* dec_W   = (const float*)d_in[18];
  const float* dec_b   = (const float*)d_in[19];
  float* outF = (float*)d_out;
  float* ctxF = outF + (size_t)B_ * F_ * NJ3_;       // motion_context base

  // ---- workspace carve (all 256B aligned) ----
  size_t off = 0;
  auto carve = [&](size_t bytes) -> void* {
    void* p = (char*)d_ws + off;
    off += (bytes + 255) & ~(size_t)255;
    return p;
  };
  unsigned short* W0a    = (unsigned short*)carve((size_t)2048 * 512 * 2);
  unsigned short* W0cat  = (unsigned short*)carve((size_t)2048 * 576 * 2);
  unsigned short* W1cat  = (unsigned short*)carve((size_t)2048 * 1024 * 2);
  unsigned short* embWt  = (unsigned short*)carve((size_t)512 * 64 * 2);
  unsigned short* niW1t  = (unsigned short*)carve((size_t)512 * 96 * 2);
  unsigned short* niW2t  = (unsigned short*)carve((size_t)1024 * 512 * 2);
  unsigned short* niW3t  = (unsigned short*)carve((size_t)2048 * 1024 * 2);
  unsigned short* decWt  = (unsigned short*)carve((size_t)64 * 512 * 2);
  float*          bsum0  = (float*)carve(2048 * 4);
  float*          bsum1  = (float*)carve(2048 * 4);
  float*          decb64 = (float*)carve(64 * 4);
  unsigned short* Xbf    = (unsigned short*)carve((size_t)65536 * 64 * 2);
  unsigned short* Xemb   = (unsigned short*)carve((size_t)65536 * 512 * 2);
  float*          XPbuf  = (float*)carve((size_t)TC_ * 256 * 2048 * 4);
  unsigned short* initbf = (unsigned short*)carve((size_t)256 * 96 * 2);
  unsigned short* z1     = (unsigned short*)carve((size_t)256 * 512 * 2);
  unsigned short* z2     = (unsigned short*)carve((size_t)256 * 1024 * 2);
  float*          z3     = (float*)carve((size_t)256 * 2048 * 4);
  unsigned short* inp0   = (unsigned short*)carve((size_t)256 * 576 * 2);
  unsigned short* inp1   = (unsigned short*)carve((size_t)256 * 1024 * 2);
  unsigned short* h1dec  = (unsigned short*)carve((size_t)256 * 512 * 2);
  float*          gates0 = (float*)carve((size_t)256 * 2048 * 4);
  float*          gates1 = (float*)carve((size_t)256 * 2048 * 4);
  float*          c0buf  = (float*)carve((size_t)256 * 512 * 4);
  float*          c1buf  = (float*)carve((size_t)256 * 512 * 4);
  float*          predb  = (float*)carve((size_t)256 * 64 * 4);
  (void)ws_size; (void)in_sizes; (void)n_in; (void)out_size;

  // ---- weight conversion to bf16 (padded / concatenated) ----
  EW(k_convT, 512 * 64,    embed_W, embWt, 34,  512,  64,  512);
  EW(k_convT, 512 * 96,    ni_W1,   niW1t, 85,  512,  96,  512);
  EW(k_convT, 1024 * 512,  ni_W2,   niW2t, 512, 1024, 512, 1024);
  EW(k_convT, 2048 * 1024, ni_W3,   niW3t, 1024,2048, 1024,2048);
  EW(k_convT, 64 * 512,    dec_W,   decWt, 512, 51,   512, 64);
  EW(k_w0a,   2048 * 512,  Wih0, W0a);
  EW(k_w0cat, 2048 * 576,  Wih0, Whh0, W0cat);
  EW(k_w1cat, 2048 * 1024, Wih1, Whh1, W1cat);
  EW(k_bias,  2048, bih0, bhh0, bih1, bhh1, dec_b, bsum0, bsum1, decb64);
  EW(k_xbf,   65536 * 64, x, Xbf);
  EW(k_initbf, 256 * 96, init_, initbf);
  EW(k_inp0init, 256 * 64, init_, inp0);

  // ---- x embedding: Xemb[f*B+b][512] (bf16) = Xbf @ embed_W + embed_b ----
  k_gemm<2, false, true, false, false, true>
      <<<GEMM_GRID(65536, 512, 2), 128, 0, stream>>>(
      Xbf, embWt, nullptr, embed_b, nullptr, Xemb, 65536, 512, 64);

  // ---- init MLP -> h0/h1/c0/c1 ----
  k_gemm<1, false, true, true, false, true>
      <<<GEMM_GRID(256, 512, 1), 128, 0, stream>>>(
      initbf, niW1t, nullptr, ni_b1, nullptr, z1, 256, 512, 96);
  k_gemm<1, false, true, true, false, true>
      <<<GEMM_GRID(256, 1024, 1), 128, 0, stream>>>(
      z1, niW2t, nullptr, ni_b2, nullptr, z2, 256, 1024, 512);
  k_gemm<1, false, true, false, true, false>
      <<<GEMM_GRID(256, 2048, 1), 128, 0, stream>>>(
      z2, niW3t, nullptr, ni_b3, z3, nullptr, 256, 2048, 1024);
  EW(k_scatter_init, 256 * 512, z3, inp0, inp1, c0buf, c1buf);

  // ---- recurrent scan, time-chunked so XP slab + weights live in L2 ----
  for (int t0 = 0; t0 < F_; t0 += TC_) {
    // XP[tl][b][2048] = Xemb[t0+tl][b] @ Wih0[:, :512].T + (bih0 + bhh0)
    k_gemm<2, false, true, false, true, false>
        <<<GEMM_GRID(TC_ * 256, 2048, 2), 128, 0, stream>>>(
        Xemb + (size_t)t0 * 256 * 512, W0a, nullptr, bsum0, XPbuf, nullptr,
        TC_ * 256, 2048, 512);

    for (int t = t0; t < t0 + TC_; ++t) {
      // layer0 gates = XP[t] + [pred|h0] @ [Wih0b|Whh0].T
      k_gemm<1, true, false, false, true, false>
          <<<GEMM_GRID(256, 2048, 1), 128, 0, stream>>>(
          inp0, W0cat, XPbuf + (size_t)(t - t0) * 256 * 2048, nullptr,
          gates0, nullptr, 256, 2048, 576);
      EW(k_cell, 256 * 512, gates0, c0buf, inp1, 1024, inp0 + 64, 576,
         (float*)nullptr, t);

      // layer1 gates = [h0n|h1] @ [Wih1|Whh1].T + (bih1 + bhh1)
      k_gemm<1, false, true, false, true, false>
          <<<GEMM_GRID(256, 2048, 1), 128, 0, stream>>>(
          inp1, W1cat, nullptr, bsum1, gates1, nullptr, 256, 2048, 1024);
      EW(k_cell, 256 * 512, gates1, c1buf, h1dec, 512, inp1 + 512, 1024,
         ctxF, t);

      // decoder: pred = h1n @ dec_W + dec_b
      k_gemm<1, false, true, false, true, false>
          <<<GEMM_GRID(256, 64, 1), 128, 0, stream>>>(
          h1dec, decWt, nullptr, decb64, predb, nullptr, 256, 64, 512);
      EW(k_spred, 256 * NJ3_, predb, outF, ctxF, inp0, t);
    }
  }
}
